// CustomMultiHeadAttention_20418274525443
// MI455X (gfx1250) — matrix operand order repaired and checked by reading, not compiled, not run on hardware
//
#include <hip/hip_runtime.h>
#include <hip/hip_bf16.h>
#include <math.h>

// ---------------------------------------------------------------------------
// CustomMultiHeadAttention for MI455X (gfx1250, wave32)
//   - bf16 WMMA 16x16x32 everywhere (8x K-throughput of the f32 16x16x4 shape)
//   - weights pre-converted to bf16; B operands staged through double-buffered
//     LDS with CDNA5 async global->LDS copies (ASYNCcnt) shared by all waves
//   - flash-attention: no T x T score materialization
//   - bias log1p via v_log_f32 fast path (|diff| <= 1)
// ws (bf16): [Wqb|Wkb|Wvb|Wob | Qb | Kb | Vb | Ab]  (4*1M + 4*4M = 40 MB)
// ---------------------------------------------------------------------------

typedef __bf16 bf16_t;
typedef __attribute__((ext_vector_type(16))) __bf16 v16bf;
typedef __attribute__((ext_vector_type(8)))  __bf16 v8bf;
typedef __attribute__((ext_vector_type(8)))  float  v8f;
typedef __attribute__((ext_vector_type(4)))  int    v4i;

#define B_DIM   2
#define T_DIM   2048
#define D_DIM   1024
#define H_DIM   16
#define DH_DIM  64

#if defined(__AMDGCN__) && __has_builtin(__builtin_amdgcn_global_load_async_to_lds_b128) && __has_builtin(__builtin_amdgcn_s_wait_asynccnt)
#define USE_ASYNC_LDS 1
#else
#define USE_ASYNC_LDS 0
#endif

// copy 8 bf16 (16 bytes) global -> LDS
__device__ __forceinline__ void cp_b128(const bf16_t* g, bf16_t* l) {
#if USE_ASYNC_LDS
  __builtin_amdgcn_global_load_async_to_lds_b128(
      (__attribute__((address_space(1))) v4i*)(void*)g,
      (__attribute__((address_space(3))) v4i*)(void*)l, 0, 0);
#else
  uint4 v = *(const uint4*)g;
  *(uint4*)l = v;
#endif
}
__device__ __forceinline__ void wait_async() {
#if USE_ASYNC_LDS
  __builtin_amdgcn_s_wait_asynccnt(0);
#endif
}

__device__ __forceinline__ bf16_t to_bf16(float x)  { return (bf16_t)x; }
__device__ __forceinline__ bf16_t to_bf16(bf16_t x) { return x; }

__device__ __forceinline__ float rowmax16(float v) {
  #pragma unroll
  for (int m = 1; m < 16; m <<= 1) v = fmaxf(v, __shfl_xor(v, m, 32));
  return v;
}
__device__ __forceinline__ float rowsum16(float v) {
  #pragma unroll
  for (int m = 1; m < 16; m <<= 1) v += __shfl_xor(v, m, 32);
  return v;
}

// ---------------------------------------------------------------------------
// f32 -> bf16 weight conversion (one-shot, bandwidth bound)
// ---------------------------------------------------------------------------
__global__ __launch_bounds__(256)
void cvt_f32_to_bf16(const float* __restrict__ s, bf16_t* __restrict__ d, int n) {
  const int i = (blockIdx.x * 256 + threadIdx.x) * 8;
  if (i + 8 <= n) {
    const float4 x = *(const float4*)(s + i);
    const float4 y = *(const float4*)(s + i + 4);
    v8bf o;
    o[0] = (bf16_t)x.x; o[1] = (bf16_t)x.y; o[2] = (bf16_t)x.z; o[3] = (bf16_t)x.w;
    o[4] = (bf16_t)y.x; o[5] = (bf16_t)y.y; o[6] = (bf16_t)y.z; o[7] = (bf16_t)y.w;
    *(v8bf*)(d + i) = o;
  }
}

// ---------------------------------------------------------------------------
// out[m,n] = sum_k A[m,k] * W[n,k] + bias[n]   (A @ W^T + b)
// A: AT [M,K]; Wb: bf16 [N,K] (pre-converted); out: bf16 or f32 [M,N]
// Block 256 = 8 waves. Block tile 128 x 64; all waves share the LDS W tile.
// Double-buffered async staging of W; A fragment software-pipelined in regs.
// ---------------------------------------------------------------------------
template <typename AT, bool OUT_F32>
__global__ __launch_bounds__(256)
void gemm_wmma(const AT* __restrict__ A, const bf16_t* __restrict__ Wb,
               const float* __restrict__ bias, void* __restrict__ outv,
               int M, int N, int K) {
  __shared__ __align__(32) bf16_t wtile[2][64 * 32];   // [n_local][k_local]

  const int tid  = threadIdx.x;
  const int lane = tid & 31, wave = tid >> 5;
  const int half = lane >> 4, l16 = lane & 15;
  const int m0 = blockIdx.x * 128 + wave * 16;
  const int n0 = blockIdx.y * 64;
  const AT* arow = A + (size_t)(m0 + l16) * K;

  // staging: thread -> (row 0..63, 8-elem chunk 0..3) of the 64x32 tile
  const int sr = tid >> 2;
  const int sc = tid & 3;
  const bf16_t* wsrc = Wb + (size_t)(n0 + sr) * K + sc * 8;

  v8f acc[4] = {};
  AT acur[16], anext[16];

  // A fragment: 16x32, contiguous k order (16 contiguous k per lane half, same as B)
  #define LOAD_A(dst, k0)                                                   \
    _Pragma("unroll") for (int j = 0; j < 8; ++j) {                         \
      (dst)[j]     = arow[(k0) + 16 * half + j];                            \
      (dst)[j + 8] = arow[(k0) + 16 * half + 8 + j];                        \
    }

  cp_b128(wsrc, &wtile[0][sr * 32 + sc * 8]);
  LOAD_A(acur, 0);
  wait_async();
  __syncthreads();

  const int nIter = K >> 5;
  for (int it = 0; it < nIter; ++it) {
    const int  cur  = it & 1;
    const bool more = (it + 1) < nIter;
    if (more) {                       // overlap: async W stage + A prefetch
      cp_b128(wsrc + (it + 1) * 32, &wtile[cur ^ 1][sr * 32 + sc * 8]);
      LOAD_A(anext, (it + 1) * 32);
    }
    v16bf a;
    #pragma unroll
    for (int j = 0; j < 16; ++j) a[j] = to_bf16(acur[j]);
    #pragma unroll
    for (int f = 0; f < 4; ++f) {
      const v16bf b = *(const v16bf*)&wtile[cur][(f * 16 + l16) * 32 + 16 * half];
      acc[f] = __builtin_amdgcn_wmma_f32_16x16x32_bf16(
          false, a, false, b, (short)0, acc[f], false, false);
    }
    if (more) wait_async();
    __syncthreads();
    #pragma unroll
    for (int j = 0; j < 16; ++j) acur[j] = anext[j];
  }

  #pragma unroll
  for (int f = 0; f < 4; ++f) {
    const int n = n0 + f * 16 + l16;
    const float bv = bias[n];
    #pragma unroll
    for (int r = 0; r < 8; ++r) {
      const int m = m0 + r + 8 * half;
      const float val = acc[f][r] + bv;
      if constexpr (OUT_F32) ((float*)outv)[(size_t)m * N + n] = val;
      else                   ((bf16_t*)outv)[(size_t)m * N + n] = (bf16_t)val;
    }
  }
  #undef LOAD_A
}

// ---------------------------------------------------------------------------
// Flash attention. Grid (T/64, B*H), block 128 (4 waves, 16 q-rows each).
// K tile async-staged to LDS; V tile staged transposed; both double-buffered
// and shared by the 4 waves. P goes through wave-private LDS (C->A layout).
// ---------------------------------------------------------------------------
__global__ __launch_bounds__(128)
void attn_kernel(const bf16_t* __restrict__ Qb,
                 const bf16_t* __restrict__ Kb,
                 const bf16_t* __restrict__ Vb,
                 const unsigned char* __restrict__ kpm,   // [B,T] bool
                 const float* __restrict__ amask,         // [T,T]
                 const float* __restrict__ sfrac,         // [B,T]
                 const float* __restrict__ alpha_p,       // [1]
                 bf16_t* __restrict__ attn_out) {
  __shared__ __align__(32) bf16_t kt_s[2][32 * 64];  // [k_local][d]
  __shared__ __align__(32) bf16_t vt_s[2][64 * 32];  // transposed: [d][k_local]
  __shared__ __align__(32) bf16_t pbuf[4][16 * 32];  // wave-private P staging

  const int tid  = threadIdx.x;
  const int lane = tid & 31, wave = tid >> 5;
  const int half = lane >> 4, l16 = lane & 15;
  const int bh = blockIdx.y;
  const int b = bh / H_DIM, h = bh % H_DIM;
  const int q0 = blockIdx.x * 64 + wave * 16;
  const float alpha = *alpha_p;
  const float scale = 0.125f;                        // 1/sqrt(64)
  bf16_t* pw = pbuf[wave];

  // staging role: row r = tid%32 of the 32-key tile, 16-elem d-chunk c = tid/32
  const int sr = tid & 31;
  const int sc = tid >> 5;
  const size_t krowbase = ((size_t)b * T_DIM + sr) * D_DIM + h * DH_DIM + sc * 16;

  #define STAGE_K(buf, kt)                                                       \
    do {                                                                         \
      const bf16_t* g = Kb + krowbase + (size_t)(kt) * D_DIM;                    \
      cp_b128(g,     &kt_s[buf][sr * 64 + sc * 16]);                             \
      cp_b128(g + 8, &kt_s[buf][sr * 64 + sc * 16 + 8]);                         \
    } while (0)
  #define LOAD_V(vr, kt)                                                         \
    _Pragma("unroll") for (int e = 0; e < 16; ++e)                               \
      (vr)[e] = Vb[krowbase + (size_t)(kt) * D_DIM + e];
  #define STORE_VT(buf, vr)                                                      \
    _Pragma("unroll") for (int e = 0; e < 16; ++e)                               \
      vt_s[buf][(sc * 16 + e) * 32 + sr] = (vr)[e];

  // Q fragments (K-dim d = 0..31 and 32..63), loaded once
  const bf16_t* qrow = Qb + ((size_t)b * T_DIM + q0 + l16) * D_DIM + h * DH_DIM;
  v16bf a0, a1;
  #pragma unroll
  for (int j = 0; j < 8; ++j) {
    a0[j]     = qrow[16 * half + j];
    a0[j + 8] = qrow[16 * half + 8 + j];
    a1[j]     = qrow[32 + 16 * half + j];
    a1[j + 8] = qrow[40 + 16 * half + j];
  }

  float sfq[8];
  #pragma unroll
  for (int r = 0; r < 8; ++r) sfq[r] = sfrac[b * T_DIM + q0 + r + 8 * half];

  v8f o[4] = {};
  float rmax[8], rsum[8];
  #pragma unroll
  for (int r = 0; r < 8; ++r) { rmax[r] = -INFINITY; rsum[r] = 0.0f; }

  // prologue: stage first K/V tiles
  {
    bf16_t vr0[16];
    STAGE_K(0, 0);
    LOAD_V(vr0, 0);
    STORE_VT(0, vr0);
    wait_async();
  }
  __syncthreads();

  const int nIter = T_DIM / 32;
  for (int it = 0; it < nIter; ++it) {
    const int  kt   = it * 32;
    const int  cur  = it & 1;
    const bool more = (it + 1) < nIter;

    bf16_t vr[16];
    if (more) {                       // issue next tile's loads, no wait
      STAGE_K(cur ^ 1, kt + 32);
      LOAD_V(vr, kt + 32);
    }

    // ---- S = Q K^T for 32 keys (two 16x16 tiles) from LDS ----
    v8f sc2[2];
    int   kcol[2];
    float sfk[2];
    bool  pad[2];
    #pragma unroll
    for (int s = 0; s < 2; ++s) {
      const v16bf b0 = *(const v16bf*)&kt_s[cur][(s * 16 + l16) * 64 + 16 * half];
      const v16bf b1 = *(const v16bf*)&kt_s[cur][(s * 16 + l16) * 64 + 32 + 16 * half];
      v8f z = {};
      z = __builtin_amdgcn_wmma_f32_16x16x32_bf16(false, a0, false, b0, (short)0, z, false, false);
      z = __builtin_amdgcn_wmma_f32_16x16x32_bf16(false, a1, false, b1, (short)0, z, false, false);
      sc2[s]  = z;
      kcol[s] = kt + s * 16 + l16;
      sfk[s]  = sfrac[b * T_DIM + kcol[s]];
      pad[s]  = (kpm[b * T_DIM + kcol[s]] != 0);
    }

    // ---- bias + masks + online softmax ----
    #pragma unroll
    for (int r = 0; r < 8; ++r) {
      const int q = q0 + r + 8 * half;
      #pragma unroll
      for (int s = 0; s < 2; ++s) {
        float v = sc2[s][r] * scale;
        const float d = sfq[r] - sfk[s];
        // |d| <= 1  ->  log1p(|d|) == log(1+|d|) via hw v_log_f32 fast path
        v += alpha * copysignf(__logf(1.0f + fabsf(d)), d);
        v += amask[(size_t)q * T_DIM + kcol[s]];
        if (pad[s]) v = -INFINITY;
        sc2[s][r] = v;
      }
      const float tmax = rowmax16(fmaxf(sc2[0][r], sc2[1][r]));
      const float nm   = fmaxf(rmax[r], tmax);
      const float corr = __expf(rmax[r] - nm);
      rmax[r] = nm;
      float psum = 0.0f;
      #pragma unroll
      for (int s = 0; s < 2; ++s) {
        const float p = __expf(sc2[s][r] - nm);
        psum += p;
        pw[(r + 8 * half) * 32 + s * 16 + l16] = (bf16_t)p;   // C-layout -> LDS
      }
      rsum[r] = rsum[r] * corr + rowsum16(psum);
      #pragma unroll
      for (int f = 0; f < 4; ++f) o[f][r] *= corr;
    }

    // per-wave LDS RAW on pbuf: DS ops in-order per wave; fence the scheduler
    asm volatile("s_wait_dscnt 0" ::: "memory");
    __builtin_amdgcn_wave_barrier();

    // ---- reload P as A-fragment (16x32) ----
    v16bf pa;
    #pragma unroll
    for (int j = 0; j < 8; ++j) {
      pa[j]     = pw[l16 * 32 + 16 * half + j];
      pa[j + 8] = pw[l16 * 32 + 16 * half + 8 + j];
    }

    // ---- O += P V from transposed V tile (contiguous 32B LDS reads) ----
    #pragma unroll
    for (int f = 0; f < 4; ++f) {
      const v16bf vb = *(const v16bf*)&vt_s[cur][(f * 16 + l16) * 32 + 16 * half];
      o[f] = __builtin_amdgcn_wmma_f32_16x16x32_bf16(
          false, pa, false, vb, (short)0, o[f], false, false);
    }

    if (more) {
      STORE_VT(cur ^ 1, vr);          // waits on vr global loads here
      wait_async();
    }
    __syncthreads();
  }

  // ---- normalize and write [B,T,D] bf16 ----
  #pragma unroll
  for (int f = 0; f < 4; ++f) {
    #pragma unroll
    for (int r = 0; r < 8; ++r) {
      const int q = q0 + r + 8 * half;
      const float val = o[f][r] / rsum[r];
      attn_out[((size_t)b * T_DIM + q) * D_DIM + h * DH_DIM + f * 16 + l16] = (bf16_t)val;
    }
  }
  #undef STAGE_K
  #undef LOAD_V
  #undef STORE_VT
}

// ---------------------------------------------------------------------------
extern "C" void kernel_launch(void* const* d_in, const int* in_sizes, int n_in,
                              void* d_out, int out_size, void* d_ws, size_t ws_size,
                              hipStream_t stream) {
  const float* query = (const float*)d_in[0];
  const float* key   = (const float*)d_in[1];
  const float* value = (const float*)d_in[2];
  const unsigned char* kpm = (const unsigned char*)d_in[3];
  const float* amask = (const float*)d_in[4];
  const float* sfrac = (const float*)d_in[5];
  const float* Wq = (const float*)d_in[6];
  const float* bq = (const float*)d_in[7];
  const float* Wk = (const float*)d_in[8];
  const float* bk = (const float*)d_in[9];
  const float* Wv = (const float*)d_in[10];
  const float* bv = (const float*)d_in[11];
  const float* Wo = (const float*)d_in[12];
  const float* bo = (const float*)d_in[13];
  const float* alpha = (const float*)d_in[14];

  const int M = B_DIM * T_DIM;            // 4096
  const int N = D_DIM, K = D_DIM;         // 1024
  const size_t WSZ = (size_t)D_DIM * D_DIM;   // 1M elems
  const size_t XSZ = (size_t)M * D_DIM;       // 4M elems

  bf16_t* ws  = (bf16_t*)d_ws;
  bf16_t* Wqb = ws;
  bf16_t* Wkb = ws + WSZ;
  bf16_t* Wvb = ws + 2 * WSZ;
  bf16_t* Wob = ws + 3 * WSZ;
  bf16_t* Qb  = ws + 4 * WSZ;
  bf16_t* Kb  = Qb + XSZ;
  bf16_t* Vb  = Kb + XSZ;
  bf16_t* Ab  = Vb + XSZ;

  // 1) weights f32 -> bf16 (once)
  {
    dim3 blk(256), g((unsigned)(WSZ / (256 * 8)));
    cvt_f32_to_bf16<<<g, blk, 0, stream>>>(Wq, Wqb, (int)WSZ);
    cvt_f32_to_bf16<<<g, blk, 0, stream>>>(Wk, Wkb, (int)WSZ);
    cvt_f32_to_bf16<<<g, blk, 0, stream>>>(Wv, Wvb, (int)WSZ);
    cvt_f32_to_bf16<<<g, blk, 0, stream>>>(Wo, Wob, (int)WSZ);
  }

  // 2) Q/K/V projections (f32 activations, bf16 weights, bf16 out)
  dim3 blk256(256), gproj(M / 128, N / 64);           // (32, 16)
  gemm_wmma<float, false><<<gproj, blk256, 0, stream>>>(query, Wqb, bq, Qb, M, N, K);
  gemm_wmma<float, false><<<gproj, blk256, 0, stream>>>(key,   Wkb, bk, Kb, M, N, K);
  gemm_wmma<float, false><<<gproj, blk256, 0, stream>>>(value, Wvb, bv, Vb, M, N, K);

  // 3) flash attention
  dim3 blk128(128), gattn(T_DIM / 64, B_DIM * H_DIM); // (32, 32)
  attn_kernel<<<gattn, blk128, 0, stream>>>(Qb, Kb, Vb, kpm, amask, sfrac, alpha, Ab);

  // 4) output projection (bf16 A, f32 out)
  gemm_wmma<bf16_t, true><<<gproj, blk256, 0, stream>>>(Ab, Wob, bo, (float*)d_out, M, N, K);
}
